// Attention_4174708212176
// MI455X (gfx1250) — matrix operand order repaired and checked by reading, not compiled, not run on hardware
//
#include <hip/hip_runtime.h>

// ---------------------------------------------------------------------------
// Types
// ---------------------------------------------------------------------------
typedef unsigned short ushort_t;
typedef __attribute__((ext_vector_type(4)))  unsigned int v4u;
typedef __attribute__((ext_vector_type(8)))  float        v8f;
typedef __attribute__((ext_vector_type(8)))  __bf16       v8bf;
typedef __attribute__((ext_vector_type(16))) __bf16       v16bf;

#define B_DIM   8
#define S_ENC   2048
#define S_DEC   2048
#define HDIM    512      // ENC_H == DEC_H == 512
#define BM      32       // decoder rows per attention workgroup

// ---------------------------------------------------------------------------
// Helpers
// ---------------------------------------------------------------------------
__device__ __forceinline__ __bf16 cvt_bf16(float f) { return (__bf16)f; }
__device__ __forceinline__ ushort_t bf_bits(__bf16 h) {
    return __builtin_bit_cast(ushort_t, h);
}
__device__ __forceinline__ float bf2f(ushort_t u) {
    return (float)__builtin_bit_cast(__bf16, u);
}
__device__ __forceinline__ v8bf load8bf(const ushort_t* p) {
    v4u t = *(const v4u*)p;           // single b128 load (global or LDS)
    return __builtin_bit_cast(v8bf, t);
}

// A fragment (16x32 bf16, M x K): lane l -> row m = l%16, half = l/16.
// Elements 0..7  hold K = half*8 .. +7 ; 8..15 hold K = 16+half*8 .. +7.
__device__ __forceinline__ v16bf fragA_bf(const ushort_t* base, int stride) {
    int l = threadIdx.x & 31;
    int m = l & 15, half = (l >> 4) & 1;
    const ushort_t* p = base + (size_t)m * stride + half * 8;
    v8bf lo = load8bf(p);
    v8bf hi = load8bf(p + 16);
    v16bf r;
#pragma unroll
    for (int i = 0; i < 8; ++i) { r[i] = lo[i]; r[i + 8] = hi[i]; }
    return r;
}

// B fragment (32x16 bf16, K x N) from an N-major (transposed) matrix:
// lane l -> col n = l%16, half = l/16; elements 0..7 hold K = half*8 .. +7 ; 8..15 hold K = 16+half*8 .. +7.
__device__ __forceinline__ v16bf fragB(const ushort_t* base, int stride) {
    int l = threadIdx.x & 31;
    int n = l & 15, half = (l >> 4) & 1;
    const ushort_t* p = base + (size_t)n * stride + half * 8;
    v8bf lo = load8bf(p);
    v8bf hi = load8bf(p + 16);
    v16bf r;
#pragma unroll
    for (int i = 0; i < 8; ++i) { r[i] = lo[i]; r[i + 8] = hi[i]; }
    return r;
}

__device__ __forceinline__ v8f wmma_bf16(v16bf a, v16bf b, v8f c) {
    return __builtin_amdgcn_wmma_f32_16x16x32_bf16(
        false, a, false, b, (short)0, c, false, false);
}

// ---------------------------------------------------------------------------
// Kernel 1: fp32 -> bf16 prep.
//   dec_bf [b][t][d] = dec ; enc_bf [b][s][e] = enc ;
//   encT [b][e][s] = enc^T ; Wt [d][e] = W^T
// ---------------------------------------------------------------------------
__global__ __launch_bounds__(256) void prep_kernel(
    const float* __restrict__ enc, const float* __restrict__ dec,
    const float* __restrict__ W,
    ushort_t* __restrict__ dec_bf, ushort_t* __restrict__ enc_bf,
    ushort_t* __restrict__ encT, ushort_t* __restrict__ Wt) {
    size_t i = (size_t)blockIdx.x * blockDim.x + threadIdx.x;
    const size_t N = (size_t)B_DIM * S_ENC * HDIM;
    if (i < N) {
        dec_bf[i] = bf_bits(cvt_bf16(dec[i]));
        enc_bf[i] = bf_bits(cvt_bf16(enc[i]));
        size_t b = i / ((size_t)HDIM * S_ENC);
        size_t r = i % ((size_t)HDIM * S_ENC);
        size_t e = r / S_ENC, s = r % S_ENC;
        encT[i] = bf_bits(cvt_bf16(enc[(b * S_ENC + s) * HDIM + e]));
    }
    if (i < (size_t)HDIM * HDIM) {
        size_t d = i >> 9, e = i & 511;
        Wt[i] = bf_bits(cvt_bf16(W[e * HDIM + d]));
    }
}

// ---------------------------------------------------------------------------
// Kernel 2: enc_proj[b] = enc[b] @ W  (bf16). Wave: 16(M)x64(N), K=16x32,
// software-pipelined B loads; A fragments are pure bf16 b128 loads.
// ---------------------------------------------------------------------------
__global__ __launch_bounds__(128) void proj_kernel(
    const ushort_t* __restrict__ enc_bf, const ushort_t* __restrict__ Wt,
    ushort_t* __restrict__ enc_proj) {
    int blk = blockIdx.x;
    int b  = blk >> 8;
    int r  = blk & 255;
    int tm = r >> 3;
    int tn = r & 7;
    int w  = threadIdx.x >> 5;

    int m0 = tm * 64 + w * 16;
    int n0 = tn * 64;
    const ushort_t* Abase = enc_bf + ((size_t)b * S_ENC + m0) * HDIM;

    v8f acc[4] = {};
    v16bf bcur[4];
#pragma unroll
    for (int j = 0; j < 4; ++j)
        bcur[j] = fragB(Wt + (size_t)(n0 + j * 16) * HDIM, HDIM);

#pragma unroll 4
    for (int kt = 0; kt < 16; ++kt) {
        v16bf a = fragA_bf(Abase + kt * 32, HDIM);
        v16bf bnext[4];
#pragma unroll
        for (int j = 0; j < 4; ++j)
            bnext[j] = (kt < 15)
                ? fragB(Wt + (size_t)(n0 + j * 16) * HDIM + (kt + 1) * 32, HDIM)
                : bcur[j];
#pragma unroll
        for (int j = 0; j < 4; ++j) acc[j] = wmma_bf16(a, bcur[j], acc[j]);
#pragma unroll
        for (int j = 0; j < 4; ++j) bcur[j] = bnext[j];
    }

    int l = threadIdx.x & 31;
    int n = l & 15, half = (l >> 4) & 1;
#pragma unroll
    for (int j = 0; j < 4; ++j)
#pragma unroll
        for (int v = 0; v < 8; ++v) {
            int m = m0 + v + 8 * half;
            int ng = n0 + j * 16 + n;
            enc_proj[((size_t)b * S_ENC + m) * HDIM + ng] =
                bf_bits(cvt_bf16(acc[j][v]));
        }
}

// ---------------------------------------------------------------------------
// Kernel 3: fused attention, BM=32 rows / WG, LDS-staged B operands.
//   LDS: Sb[32][2048] bf16 (128K) | Qs[32][512] bf16 (32K) | stage 2x64K
//   8 waves: mi = w>>2 (row half), si/ei = w&3 (col quarter).
//   Phase 1: S = Q @ Kp^T, enc_proj staged per 64-row chunk, double-buffered.
//   Phase 2: softmax in LDS (4 rows per wave), normalized bf16 written back.
//   Phase 3: O = P @ V, encT staged per 32-col chunk [512e][32s], dbl-buffered.
// ---------------------------------------------------------------------------
__global__ __launch_bounds__(256) void attn_kernel(
    const ushort_t* __restrict__ dec_bf, const ushort_t* __restrict__ encT,
    const ushort_t* __restrict__ enc_proj, float* __restrict__ out) {
    extern __shared__ char smem[];
    ushort_t* Sb  = (ushort_t*)smem;                         // [32][2048]
    ushort_t* Qs  = (ushort_t*)(smem + 131072);              // [32][512]
    ushort_t* St0 = (ushort_t*)(smem + 131072 + 32768);      // 64KB
    ushort_t* St1 = St0 + 32768;                             // 64KB

    int b   = blockIdx.x >> 6;                 // 64 row-blocks per batch
    int t0  = (blockIdx.x & 63) * BM;
    int w   = threadIdx.x >> 5;
    int l   = threadIdx.x & 31;
    int n   = l & 15, half = (l >> 4) & 1;
    int tid = threadIdx.x;
    int mi  = w >> 2;                          // 0..1 : row sub-block
    int qi  = w & 3;                           // 0..3 : col quarter

    // ---- Load the 32x512 Q slab into LDS (flat b128 copy) ----
    {
        const ushort_t* src = dec_bf + ((size_t)b * S_DEC + t0) * HDIM;
#pragma unroll
        for (int i = 0; i < 8; ++i) {
            int g = tid + i * 256;             // 2048 groups of 8 ushorts
            *(v4u*)(Qs + g * 8) = *(const v4u*)(src + g * 8);
        }
    }

    const ushort_t* Qa = Qs + (size_t)(mi * 16) * HDIM;
    const ushort_t* Kp = enc_proj + (size_t)b * S_ENC * HDIM;

    // ---- Phase 1 prologue: stage chunk 0 (64 s-rows x 512 d = 64KB) ----
#pragma unroll
    for (int i = 0; i < 16; ++i) {
        int g = tid + i * 256;                 // 4096 groups of 8
        *(v4u*)(St0 + g * 8) = *(const v4u*)(Kp + g * 8);
    }
    __syncthreads();

    for (int c = 0; c < 32; ++c) {
        ushort_t* cur = (c & 1) ? St1 : St0;
        ushort_t* nxt = (c & 1) ? St0 : St1;

        // issue next chunk's global loads (overlap with WMMAs below)
        v4u creg[16];
        if (c < 31) {
            const ushort_t* src = Kp + (size_t)(c + 1) * 64 * HDIM;
#pragma unroll
            for (int i = 0; i < 16; ++i)
                creg[i] = *(const v4u*)(src + (tid + i * 256) * 8);
        }

        // compute this wave's 16x16 logits tile over K=512
        v8f acc = {};
        const ushort_t* Bb = cur + (size_t)(qi * 16) * HDIM;
#pragma unroll
        for (int kt = 0; kt < 16; ++kt) {
            v16bf a  = fragA_bf(Qa + kt * 32, HDIM);
            v16bf bf = fragB(Bb + kt * 32, HDIM);
            acc = wmma_bf16(a, bf, acc);
        }
        int s0 = c * 64 + qi * 16;
#pragma unroll
        for (int v = 0; v < 8; ++v)
            Sb[(size_t)(mi * 16 + v + 8 * half) * S_ENC + s0 + n] =
                bf_bits(cvt_bf16(acc[v]));

        // drain staged registers into the other buffer
        if (c < 31) {
#pragma unroll
            for (int i = 0; i < 16; ++i)
                *(v4u*)(nxt + (tid + i * 256) * 8) = creg[i];
        }
        __syncthreads();
    }

    // ---- Phase 2: softmax, 4 rows per wave, bf16 in/out ----
    for (int r = w * 4; r < w * 4 + 4; ++r) {
        ushort_t* row = Sb + (size_t)r * S_ENC;
        float mx = -3.4e38f;
        for (int i = l; i < S_ENC; i += 32) mx = fmaxf(mx, bf2f(row[i]));
#pragma unroll
        for (int o = 16; o > 0; o >>= 1) mx = fmaxf(mx, __shfl_xor(mx, o, 32));
        float sum = 0.f;
        for (int i = l; i < S_ENC; i += 32) {
            float e = __expf(bf2f(row[i]) - mx);
            sum += e;
            row[i] = bf_bits(cvt_bf16(e));
        }
#pragma unroll
        for (int o = 16; o > 0; o >>= 1) sum += __shfl_xor(sum, o, 32);
        float inv = 1.0f / sum;
        for (int i = 2 * l; i < S_ENC; i += 64) {
            unsigned lo = bf_bits(cvt_bf16(bf2f(row[i]) * inv));
            unsigned hi = bf_bits(cvt_bf16(bf2f(row[i + 1]) * inv));
            *(unsigned*)(row + i) = lo | (hi << 16);     // packed b32 store
        }
    }
    __syncthreads();

    // ---- Phase 3: O = P @ V, staged encT tiles [512e][32s] (32KB each) ----
    const ushort_t* Vb = encT + (size_t)b * HDIM * S_ENC;
    const ushort_t* Pa = Sb + (size_t)(mi * 16) * S_ENC;

    // prologue: stage kt=0 tile
#pragma unroll
    for (int i = 0; i < 8; ++i) {
        int g = tid + i * 256;                 // 2048 groups, 4 per e-row
        int e = g >> 2, q = g & 3;
        *(v4u*)(St0 + e * 32 + q * 8) = *(const v4u*)(Vb + (size_t)e * S_ENC + q * 8);
    }
    __syncthreads();

    v8f oacc[8] = {};
    for (int kt = 0; kt < 64; ++kt) {
        ushort_t* cur = (kt & 1) ? St1 : St0;
        ushort_t* nxt = (kt & 1) ? St0 : St1;

        v4u creg[8];
        if (kt < 63) {
            int s0n = (kt + 1) * 32;
#pragma unroll
            for (int i = 0; i < 8; ++i) {
                int g = tid + i * 256;
                int e = g >> 2, q = g & 3;
                creg[i] = *(const v4u*)(Vb + (size_t)e * S_ENC + s0n + q * 8);
            }
        }

        v16bf a = fragA_bf(Pa + kt * 32, S_ENC);   // LDS b128 reads
#pragma unroll
        for (int j = 0; j < 8; ++j) {
            v16bf bf = fragB(cur + (size_t)(qi * 128 + j * 16) * 32, 32);
            oacc[j] = wmma_bf16(a, bf, oacc[j]);
        }

        if (kt < 63) {
#pragma unroll
            for (int i = 0; i < 8; ++i) {
                int g = tid + i * 256;
                int e = g >> 2, q = g & 3;
                *(v4u*)(nxt + e * 32 + q * 8) = creg[i];
            }
        }
        __syncthreads();
    }

    // ---- epilogue: write O (f32) ----
#pragma unroll
    for (int j = 0; j < 8; ++j)
#pragma unroll
        for (int v = 0; v < 8; ++v) {
            int t = t0 + mi * 16 + v + 8 * half;
            int e = qi * 128 + j * 16 + n;
            out[((size_t)b * S_DEC + t) * HDIM + e] = oacc[j][v];
        }
}

// ---------------------------------------------------------------------------
// Host launcher
// ---------------------------------------------------------------------------
extern "C" void kernel_launch(void* const* d_in, const int* in_sizes, int n_in,
                              void* d_out, int out_size, void* d_ws, size_t ws_size,
                              hipStream_t stream) {
    const float* enc = (const float*)d_in[0];   // [8][2048][512] f32
    const float* dec = (const float*)d_in[1];   // [8][2048][512] f32
    const float* W   = (const float*)d_in[2];   // [512][512] f32
    float* out = (float*)d_out;                 // [8][2048][512] f32

    const size_t SEG = (size_t)B_DIM * S_ENC * HDIM * sizeof(ushort_t); // 16 MiB
    char* ws = (char*)d_ws;
    ushort_t* dec_bf   = (ushort_t*)(ws);
    ushort_t* enc_bf   = (ushort_t*)(ws + SEG);
    ushort_t* encT     = (ushort_t*)(ws + 2 * SEG);
    ushort_t* enc_proj = (ushort_t*)(ws + 3 * SEG);
    ushort_t* Wt       = (ushort_t*)(ws + 4 * SEG);

    const int prep_threads = 256;
    const size_t N = (size_t)B_DIM * S_ENC * HDIM;
    const int prep_blocks = (int)((N + prep_threads - 1) / prep_threads);
    prep_kernel<<<prep_blocks, prep_threads, 0, stream>>>(
        enc, dec, W, dec_bf, enc_bf, encT, Wt);

    proj_kernel<<<2048, 128, 0, stream>>>(enc_bf, Wt, enc_proj);

    // LDS: 128K (S bf16) + 32K (Q slab) + 2x64K (stage) = 288 KB
    const size_t lds_bytes = 131072 + 32768 + 2 * 65536;
    attn_kernel<<<B_DIM * (S_DEC / BM), 256, lds_bytes, stream>>>(
        dec_bf, encT, enc_proj, out);
}
